// FlashAttention_77575699300715
// MI455X (gfx1250) — matrix operand order repaired and checked by reading, not compiled, not run on hardware
//
#include <hip/hip_runtime.h>

typedef __attribute__((ext_vector_type(16))) __bf16 v16bf;
typedef __attribute__((ext_vector_type(8)))  __bf16 v8bf;
typedef __attribute__((ext_vector_type(2)))  __bf16 v2bf;
typedef __attribute__((ext_vector_type(8)))  float  v8f;

#define B_  8
#define LQ_ 4096
#define LK_ 4096
#define D_  128
#define QT  128          // q rows per block = 8 waves x 16
#define KT  64           // keys per iteration
#define KS_STRIDE 136    // bf16 elems per K row in LDS (128 + 8 pad)
#define VT_STRIDE 72     // bf16 elems per Vt row (64 + 8 pad)
#define PS_STRIDE 72     // bf16 elems per P row (64 + 8 pad)

union FragBF  { v16bf v; v8bf h[2]; v2bf p[8]; };
union Pack8BF { v8bf  v; v2bf p[4]; };

#ifndef __has_builtin
#define __has_builtin(x) 0
#endif

__device__ inline v2bf pk_bf16(float a, float b) {
#if __has_builtin(__builtin_amdgcn_cvt_pk_bf16_f32)
    return __builtin_bit_cast(v2bf, __builtin_amdgcn_cvt_pk_bf16_f32(a, b));
#else
    v2bf r; r[0] = (__bf16)a; r[1] = (__bf16)b; return r;
#endif
}

__device__ inline v8f vzero8() {
    v8f z;
    #pragma unroll
    for (int e = 0; e < 8; ++e) z[e] = 0.0f;
    return z;
}

// Cooperative load of one K tile (row-major bf16) and one V tile (transposed,
// key-pairs packed so the transpose uses b32 stores).
__device__ inline void load_tiles(const float* __restrict__ kbase,
                                  const float* __restrict__ vbase,
                                  __bf16* KsBuf, __bf16* VtBuf,
                                  int t, bool prefetch_next)
{
    // ---- K: thread -> (row = t>>2, 32 floats at (t&3)*32)
    {
        const int lrow = t >> 2;
        const int ld0  = (t & 3) * 32;
        const float* kp = kbase + (size_t)lrow * D_ + ld0;
        v8bf* dst = (v8bf*)&KsBuf[lrow * KS_STRIDE + ld0];
        #pragma unroll
        for (int g = 0; g < 4; ++g) {              // 4 groups of 8 floats
            float4 f0 = ((const float4*)kp)[2*g + 0];
            float4 f1 = ((const float4*)kp)[2*g + 1];
            Pack8BF pk;
            pk.p[0] = pk_bf16(f0.x, f0.y);
            pk.p[1] = pk_bf16(f0.z, f0.w);
            pk.p[2] = pk_bf16(f1.x, f1.y);
            pk.p[3] = pk_bf16(f1.z, f1.w);
            dst[g] = pk.v;
        }
        if (prefetch_next) __builtin_prefetch(kp + (size_t)KT * D_, 0, 1);
    }
    // ---- V: thread -> key pair (t>>3)*2, 16 d-values at (t&7)*16; packed stores
    {
        const int kpair = t >> 3;      // 0..31
        const int vd0   = (t & 7) * 16;
        const float* va = vbase + (size_t)(2 * kpair) * D_ + vd0;
        float4 fa[4], fb[4];
        #pragma unroll
        for (int i = 0; i < 4; ++i) fa[i] = ((const float4*)va)[i];
        #pragma unroll
        for (int i = 0; i < 4; ++i) fb[i] = ((const float4*)(va + D_))[i];
        const float* faf = (const float*)&fa[0];
        const float* fbf = (const float*)&fb[0];
        #pragma unroll
        for (int i = 0; i < 16; ++i)
            *(v2bf*)&VtBuf[(vd0 + i) * VT_STRIDE + 2 * kpair] = pk_bf16(faf[i], fbf[i]);
        if (prefetch_next) __builtin_prefetch(va + (size_t)KT * D_, 0, 1);
    }
}

__global__ __launch_bounds__(256) void
fa_fwd_bf16_wmma(const float* __restrict__ q,
                 const float* __restrict__ k,
                 const float* __restrict__ v,
                 const unsigned char* __restrict__ mask,
                 float* __restrict__ out)
{
    __shared__ __bf16 Ks[2][KT * KS_STRIDE];     // double-buffered K tile [key][d]
    __shared__ __bf16 Vt[2][D_ * VT_STRIDE];     // double-buffered V tile [d][key]
    __shared__ __bf16 Ps[8 * 16 * PS_STRIDE];    // per-wave P scratch [qrow][key]

    const int t    = threadIdx.x;     // 0..255
    const int wid  = t >> 5;          // wave id 0..7
    const int lane = t & 31;
    const int half = lane >> 4;       // 0 or 1
    const int lm   = lane & 15;

    const int batch = blockIdx.y;
    const int q0    = blockIdx.x * QT + wid * 16;

    // exp2 domain: fold softmax scale * log2(e) into Q
    const float qscale = 0.08838834764831845f * 1.4426950408889634f;

    // ---- Load this wave's Q tile (16 rows x 128) into A fragments, pre-scaled.
    FragBF aq[4];
    {
        const float* qp = q + ((size_t)batch * LQ_ + (q0 + lm)) * D_;
        #pragma unroll
        for (int kc = 0; kc < 4; ++kc) {
            const int dA = kc * 32 + half * 16;
            FragBF a;
            #pragma unroll
            for (int e = 0; e < 4; ++e)
                a.p[e]     = pk_bf16(qp[dA + 2*e]      * qscale, qp[dA + 2*e + 1]      * qscale);
            #pragma unroll
            for (int e = 0; e < 4; ++e)
                a.p[4 + e] = pk_bf16(qp[dA + 8 + 2*e] * qscale, qp[dA + 8 + 2*e + 1] * qscale);
            aq[kc] = a;
        }
    }

    // constant all-ones B fragment: l-row-sum via WMMA (P x ones)
    FragBF vones;
    #pragma unroll
    for (int e = 0; e < 16; ++e) vones.v[e] = (__bf16)1.0f;

    v8f o[8];
    #pragma unroll
    for (int n = 0; n < 8; ++n) o[n] = vzero8();
    v8f l9 = vzero8();                 // running row-sum accumulator (all 16 cols equal)

    float mrun[8];
    #pragma unroll
    for (int r = 0; r < 8; ++r) mrun[r] = -3.0e38f;

    const float* kB = k + (size_t)batch * LK_ * D_;
    const float* vB = v + (size_t)batch * LK_ * D_;
    const int nkt = LK_ / KT;   // 64 iterations

    // prologue: tile 0 into buffer 0
    load_tiles(kB, vB, &Ks[0][0], &Vt[0][0], t, true);

    for (int it = 0; it < nkt; ++it) {
        const int kt0 = it * KT;
        const int cur = it & 1;
        __syncthreads();   // buf[cur] ready; buf[1-cur] consumers finished

        // issue next tile's global loads before compute (double buffer)
        if (it + 1 < nkt)
            load_tiles(kB + (size_t)(it + 1) * KT * D_,
                       vB + (size_t)(it + 1) * KT * D_,
                       &Ks[1 - cur][0], &Vt[1 - cur][0], t, it + 2 < nkt);

        const __bf16* KsC = &Ks[cur][0];
        const __bf16* VtC = &Vt[cur][0];

        auto loadKfrag = [&](int i) -> FragBF {   // i = j*4 + kc
            const int j = i >> 2, kc = i & 3;
            FragBF b;
            const __bf16* p = &KsC[(j*16 + lm) * KS_STRIDE + kc*32 + half*16];
            b.h[0] = *(const v8bf*)p;
            b.h[1] = *(const v8bf*)(p + 8);
            return b;
        };
        auto loadVfrag = [&](int i) -> FragBF {   // i = kc*8 + n
            const int kc = i >> 3, n = i & 7;
            FragBF b;
            const __bf16* p = &VtC[(n*16 + lm) * VT_STRIDE + kc*32 + half*16];
            b.h[0] = *(const v8bf*)p;
            b.h[1] = *(const v8bf*)(p + 8);
            return b;
        };

        // ---- S = (Q*qscale) K^T : depth-2 pipelined B-fragment stream
        v8f s[4];
        #pragma unroll
        for (int j = 0; j < 4; ++j) s[j] = vzero8();
        {
            FragBF f0 = loadKfrag(0);
            FragBF f1 = loadKfrag(1);
            #pragma unroll
            for (int i = 0; i < 16; ++i) {
                FragBF f2 = (i + 2 < 16) ? loadKfrag(i + 2) : f0;
                const int j = i >> 2, kc = i & 3;
                s[j] = __builtin_amdgcn_wmma_f32_16x16x32_bf16(
                           false, aq[kc].v, false, f0.v, (short)0, s[j], false, false);
                f0 = f1; f1 = f2;
            }
        }

        // ---- apply mask (True => -inf); D-frag row r maps to q row r + 8*half
        {
            const unsigned char* mb =
                mask + ((size_t)batch * LQ_ + (q0 + 8*half)) * LK_ + kt0 + lm;
            #pragma unroll
            for (int r = 0; r < 8; ++r) {
                const unsigned char* mr = mb + (size_t)r * LK_;
                #pragma unroll
                for (int j = 0; j < 4; ++j)
                    if (mr[j*16]) s[j][r] = -1.0e30f;
            }
        }

        // ---- online softmax (exp2 domain); row max via 16-lane shuffles
        float alpha[8];
        #pragma unroll
        for (int r = 0; r < 8; ++r) {
            float mx = fmaxf(fmaxf(s[0][r], s[1][r]), fmaxf(s[2][r], s[3][r]));
            mx = fmaxf(mx, __shfl_xor(mx, 1, 16));
            mx = fmaxf(mx, __shfl_xor(mx, 2, 16));
            mx = fmaxf(mx, __shfl_xor(mx, 4, 16));
            mx = fmaxf(mx, __shfl_xor(mx, 8, 16));
            const float mnew = fmaxf(mrun[r], mx);
            alpha[r] = exp2f(mrun[r] - mnew);
            mrun[r]  = mnew;

            __bf16* pr = &Ps[((wid*16) + r + 8*half) * PS_STRIDE + lm];
            #pragma unroll
            for (int j = 0; j < 4; ++j)
                pr[j*16] = (__bf16)exp2f(s[j][r] - mnew);
        }
        // rescale running O and l accumulators
        #pragma unroll
        for (int r = 0; r < 8; ++r) l9[r] *= alpha[r];
        #pragma unroll
        for (int n = 0; n < 8; ++n) {
            #pragma unroll
            for (int r = 0; r < 8; ++r)
                o[n][r] *= alpha[r];
        }

        // ---- O += P V ; l += P * ones (matrix-pipe row sum); depth-2 pipeline
        {
            FragBF pa[2];
            #pragma unroll
            for (int kc = 0; kc < 2; ++kc) {
                const __bf16* pp = &Ps[(wid*16 + lm) * PS_STRIDE + kc*32 + half*16];
                pa[kc].h[0] = *(const v8bf*)pp;
                pa[kc].h[1] = *(const v8bf*)(pp + 8);
            }
            l9 = __builtin_amdgcn_wmma_f32_16x16x32_bf16(
                     false, pa[0].v, false, vones.v, (short)0, l9, false, false);
            l9 = __builtin_amdgcn_wmma_f32_16x16x32_bf16(
                     false, pa[1].v, false, vones.v, (short)0, l9, false, false);

            FragBF g0 = loadVfrag(0);
            FragBF g1 = loadVfrag(1);
            #pragma unroll
            for (int i = 0; i < 16; ++i) {
                FragBF g2 = (i + 2 < 16) ? loadVfrag(i + 2) : g0;
                const int kc = i >> 3, n = i & 7;
                o[n] = __builtin_amdgcn_wmma_f32_16x16x32_bf16(
                           false, pa[kc].v, false, g0.v, (short)0, o[n], false, false);
                g0 = g1; g1 = g2;
            }
        }
    }

    // ---- epilogue: normalize and store fp32 output
    #pragma unroll
    for (int r = 0; r < 8; ++r) {
        const float inv = __builtin_amdgcn_rcpf(l9[r]);
        float* op = out + ((size_t)batch * LQ_ + (q0 + r + 8*half)) * D_ + lm;
        #pragma unroll
        for (int n = 0; n < 8; ++n)
            op[n*16] = o[n][r] * inv;
    }
}

extern "C" void kernel_launch(void* const* d_in, const int* in_sizes, int n_in,
                              void* d_out, int out_size, void* d_ws, size_t ws_size,
                              hipStream_t stream)
{
    const float*         q  = (const float*)d_in[0];
    const float*         k  = (const float*)d_in[1];
    const float*         v  = (const float*)d_in[2];
    const unsigned char* m  = (const unsigned char*)d_in[3];
    float*               o  = (float*)d_out;

    dim3 grid(LQ_ / QT, B_);   // 32 x 8 blocks
    dim3 block(256);           // 8 wave32
    fa_fwd_bf16_wmma<<<grid, block, 0, stream>>>(q, k, v, m, o);
}
